// CausalSelfAttention_89996744720885
// MI455X (gfx1250) — matrix operand order repaired and checked by reading, not compiled, not run on hardware
//
#include <hip/hip_runtime.h>
#include <hip/hip_bf16.h>

typedef __bf16 bf16_t;
typedef __attribute__((ext_vector_type(16))) __bf16 v16bf;
typedef __attribute__((ext_vector_type(8)))  __bf16 v8bf;
typedef __attribute__((ext_vector_type(8)))  float  v8f;
typedef __attribute__((ext_vector_type(4))) unsigned int v4u;
typedef __attribute__((ext_vector_type(8)))  int v8i;
typedef __attribute__((ext_vector_type(4)))  int v4i;

#define WMMA_BF16(a, b, c) \
  __builtin_amdgcn_wmma_f32_16x16x32_bf16(false, (a), false, (b), (short)0, (c), false, false)

// ---- fragment loaders using contiguous k = 16*(lane/16) + s for both A and B (wave32) ----
__device__ __forceinline__ v16bf make_a(const bf16_t* __restrict__ tile, int ld, int lane) {
  const int g = lane >> 4, r = lane & 15;
  const bf16_t* p = tile + r * ld + g * 16;
  v8bf lo = *(const v8bf*)(p);
  v8bf hi = *(const v8bf*)(p + 8);
  v16bf a;
#pragma unroll
  for (int i = 0; i < 8; ++i) { a[i] = lo[i]; a[i + 8] = hi[i]; }
  return a;
}
__device__ __forceinline__ v16bf make_b(const bf16_t* __restrict__ tile, int ld, int lane) {
  const int n = lane & 15, kb = (lane >> 4) * 16;
  return *(const v16bf*)(tile + n * ld + kb);
}

// ---- TDM: 2D tile load Global -> LDS (CDNA5 ISA) ----
// dims/strides/tiles in elements (data_size = 2 bytes)
__device__ __forceinline__ void tdm_load_2d(const void* gaddr, unsigned lds_off,
                                            unsigned dim0, unsigned dim1,
                                            unsigned stride0, unsigned tile0,
                                            unsigned tile1) {
  const unsigned long long ga = (unsigned long long)gaddr;
  v4u g0;
  g0[0] = 1u;                                   // count=1, user D#
  g0[1] = lds_off;                              // lds_addr[31:0]
  g0[2] = (unsigned)ga;                         // global_addr[31:0]
  g0[3] = (unsigned)(ga >> 32) | (2u << 30);    // global_addr[56:32] | type=2
  v8i g1;
  g1[0] = 0x10000;                              // workgroup_mask=0, data_size=1 (2B)
  g1[1] = (int)((dim0 & 0xFFFFu) << 16);                         // tensor_dim0[15:0] @48
  g1[2] = (int)(((dim0 >> 16) & 0xFFFFu) | ((dim1 & 0xFFFFu) << 16)); // dim0 hi, dim1 lo
  g1[3] = (int)(((dim1 >> 16) & 0xFFFFu) | (tile0 << 16));       // dim1 hi, tile_dim0
  g1[4] = (int)(tile1 & 0xFFFFu);                                // tile_dim1, tile_dim2=0
  g1[5] = (int)stride0;                                          // tensor_dim0_stride[31:0]
  g1[6] = 0;
  g1[7] = 0;
  v4i z4 = {0, 0, 0, 0};
#if defined(__clang_major__) && (__clang_major__ >= 23)
  v8i z8 = {0, 0, 0, 0, 0, 0, 0, 0};
  __builtin_amdgcn_tensor_load_to_lds(g0, g1, z4, z4, z8, 0);
#else
  __builtin_amdgcn_tensor_load_to_lds(g0, g1, z4, z4, 0);
#endif
}

// ---- f32 -> bf16 convert ----
__global__ void cvt_kernel(const float* __restrict__ src, bf16_t* __restrict__ dst, int n) {
  int i = blockIdx.x * blockDim.x + threadIdx.x;
  if (i < n) dst[i] = (bf16_t)src[i];
}

// ---- fused QKV projection + RoPE, 32x64 tile per wave ----
__global__ __launch_bounds__(128) void qkv_rope_kernel(
    const bf16_t* __restrict__ xh, const bf16_t* __restrict__ wqh,
    const bf16_t* __restrict__ wkh, const bf16_t* __restrict__ wvh,
    const float* __restrict__ cosb, const float* __restrict__ sinb,
    bf16_t* __restrict__ Qh, bf16_t* __restrict__ Kh, bf16_t* __restrict__ Vt) {
  const int lane = threadIdx.x & 31;
  const int w = blockIdx.x * 4 + (threadIdx.x >> 5);
  const int mt = w & 127;          // 128 row tiles of 32 (B*T = 4096)
  const int n64 = (w >> 7) & 15;   // 16 col tiles of 64
  const int z = w >> 11;           // 0:q 1:k 2:v
  if (z >= 3) return;
  const bf16_t* Wm = (z == 0) ? wqh : ((z == 1) ? wkh : wvh);
  const int m0 = mt * 32, n0 = n64 * 64;
  v8f acc[2][4] = {};
  for (int k0 = 0; k0 < 1024; k0 += 32) {
    v16bf a0 = make_a(xh + (size_t)m0 * 1024 + k0, 1024, lane);
    v16bf a1 = make_a(xh + (size_t)(m0 + 16) * 1024 + k0, 1024, lane);
#pragma unroll
    for (int j = 0; j < 4; ++j) {
      v16bf b = make_b(Wm + (size_t)(n0 + 16 * j) * 1024 + k0, 1024, lane);
      acc[0][j] = WMMA_BF16(a0, b, acc[0][j]);
      acc[1][j] = WMMA_BF16(a1, b, acc[1][j]);
    }
  }
  const int hi = lane >> 4, nn = lane & 15;
  if (z < 2) {  // RoPE + store (B,H,T,D)
    bf16_t* dst = (z == 0) ? Qh : Kh;
#pragma unroll
    for (int mi = 0; mi < 2; ++mi)
#pragma unroll
      for (int j = 0; j < 4; ++j)
#pragma unroll
        for (int r = 0; r < 8; ++r) {
          const int row = m0 + 16 * mi + r + 8 * hi;      // b*T + t
          const int col = n0 + 16 * j + nn;               // h*D + d
          const int b = row >> 11, t = row & 2047;
          const int h = col >> 6, d = col & 63;
          const float c = cosb[t * 32 + (d >> 1)];
          const float s = sinb[t * 32 + (d >> 1)];
          const float v = acc[mi][j][r];
          const float p = __shfl_xor(v, 1, 32);           // even/odd partner
          const float o = (d & 1) ? (p * s + v * c) : (v * c - p * s);
          dst[((size_t)(b * 16 + h) * 2048 + t) * 64 + d] = (bf16_t)o;
        }
  } else {      // V transposed: (B,H,D,T)
#pragma unroll
    for (int mi = 0; mi < 2; ++mi)
#pragma unroll
      for (int j = 0; j < 4; ++j)
#pragma unroll
        for (int r = 0; r < 8; ++r) {
          const int row = m0 + 16 * mi + r + 8 * hi;
          const int col = n0 + 16 * j + nn;
          const int b = row >> 11, t = row & 2047;
          const int h = col >> 6, d = col & 63;
          Vt[((size_t)(b * 16 + h) * 64 + d) * 2048 + t] = (bf16_t)acc[mi][j][r];
        }
  }
}

// ---- flash attention: block = 4 waves share one (b,h); K/V staged via TDM ----
__global__ __launch_bounds__(128) void attn_kernel(
    const bf16_t* __restrict__ Qh, const bf16_t* __restrict__ Kh,
    const bf16_t* __restrict__ Vt, bf16_t* __restrict__ Oh) {
  __shared__ __align__(128) bf16_t Kl[32 * 64];        // key chunk, ld 64
  __shared__ __align__(128) bf16_t Vl[64 * 32];        // V^T chunk, ld 32
  __shared__ __align__(128) bf16_t pbuf[4][16 * 32];
  const int lane = threadIdx.x & 31;
  const int wl = threadIdx.x >> 5;
  const int bh = blockIdx.x >> 5;                      // 32 (b,h) pairs
  const int qg = blockIdx.x & 31;                      // 32 query groups of 64
  const int t0 = (qg * 4 + wl) * 16;
  const int t0max = qg * 64 + 48;
  const bf16_t* Qb = Qh + (size_t)bh * 2048 * 64;
  const bf16_t* Kb = Kh + (size_t)bh * 2048 * 64;
  const bf16_t* Vb = Vt + (size_t)bh * 64 * 2048;
  const int hi = lane >> 4, nn = lane & 15;
  bf16_t* P = pbuf[wl];

  v16bf aq0 = make_a(Qb + t0 * 64 + 0, 64, lane);
  v16bf aq1 = make_a(Qb + t0 * 64 + 32, 64, lane);
  v8f acc[4] = {};
  float mrow[8], lrow[8];
#pragma unroll
  for (int r = 0; r < 8; ++r) { mrow[r] = -1e30f; lrow[r] = 0.f; }

  for (int s0 = 0; s0 <= t0max; s0 += 32) {
    if (wl == 0) {  // one wave drives the TDM for the whole block
      tdm_load_2d(Kb + (size_t)s0 * 64, (unsigned)(uintptr_t)Kl,
                  /*dim0*/64, /*dim1*/2048, /*stride0*/64, /*tile0*/64, /*tile1*/32);
      tdm_load_2d(Vb + s0, (unsigned)(uintptr_t)Vl,
                  /*dim0*/2048, /*dim1*/64, /*stride0*/2048, /*tile0*/32, /*tile1*/64);
      __builtin_amdgcn_s_wait_tensorcnt(0);
    }
    __syncthreads();
    if (s0 <= t0) {
      v8f S0 = {}, S1 = {};
      {
        v16bf b00 = make_b(Kl, 64, lane);
        v16bf b01 = make_b(Kl + 32, 64, lane);
        S0 = WMMA_BF16(aq0, b00, S0);
        S0 = WMMA_BF16(aq1, b01, S0);
        v16bf b10 = make_b(Kl + 16 * 64, 64, lane);
        v16bf b11 = make_b(Kl + 16 * 64 + 32, 64, lane);
        S1 = WMMA_BF16(aq0, b10, S1);
        S1 = WMMA_BF16(aq1, b11, S1);
      }
      const float sc = 0.125f;  // 1/sqrt(64)
      float mnew[8];
#pragma unroll
      for (int r = 0; r < 8; ++r) {
        const int q = t0 + r + 8 * hi;
        float x0 = (s0 + nn <= q) ? S0[r] * sc : -1e30f;
        float x1 = (s0 + 16 + nn <= q) ? S1[r] * sc : -1e30f;
        S0[r] = x0; S1[r] = x1;
        float mx = fmaxf(x0, x1);
#pragma unroll
        for (int msk = 1; msk < 16; msk <<= 1) mx = fmaxf(mx, __shfl_xor(mx, msk, 32));
        mnew[r] = fmaxf(mrow[r], mx);
      }
#pragma unroll
      for (int r = 0; r < 8; ++r) {
        const float p0 = __expf(S0[r] - mnew[r]);
        const float p1 = __expf(S1[r] - mnew[r]);
        const int rr = r + 8 * hi;
        P[rr * 32 + nn] = (bf16_t)p0;
        P[rr * 32 + 16 + nn] = (bf16_t)p1;
        float rs = p0 + p1;
#pragma unroll
        for (int msk = 1; msk < 16; msk <<= 1) rs += __shfl_xor(rs, msk, 32);
        const float corr = __expf(mrow[r] - mnew[r]);
        lrow[r] = lrow[r] * corr + rs;
        mrow[r] = mnew[r];
#pragma unroll
        for (int j = 0; j < 4; ++j) acc[j][r] *= corr;
      }
      asm volatile("s_wait_dscnt 0x0" ::: "memory");  // wave-sync LDS P visible
      v16bf ap = make_a(P, 32, lane);
#pragma unroll
      for (int j = 0; j < 4; ++j) {
        v16bf bv = make_b(Vl + 16 * j * 32, 32, lane);
        acc[j] = WMMA_BF16(ap, bv, acc[j]);
      }
    }
    __syncthreads();  // protect Kl/Vl before next TDM chunk
  }
  // normalize + store (B,T,C) bf16
  const int bb = bh >> 4, h = bh & 15;
#pragma unroll
  for (int j = 0; j < 4; ++j)
#pragma unroll
    for (int r = 0; r < 8; ++r) {
      const int t = t0 + r + 8 * hi;
      const int col = h * 64 + 16 * j + nn;
      Oh[((size_t)bb * 2048 + t) * 1024 + col] = (bf16_t)(acc[j][r] / lrow[r]);
    }
}

// ---- output projection: y = O @ wo^T, 32x64 tile per wave, f32 out ----
__global__ __launch_bounds__(128) void oproj_kernel(
    const bf16_t* __restrict__ Oh, const bf16_t* __restrict__ woh,
    float* __restrict__ out) {
  const int lane = threadIdx.x & 31;
  const int w = blockIdx.x * 4 + (threadIdx.x >> 5);
  const int mt = w & 127, n64 = w >> 7;
  const int m0 = mt * 32, n0 = n64 * 64;
  v8f acc[2][4] = {};
  for (int k0 = 0; k0 < 1024; k0 += 32) {
    v16bf a0 = make_a(Oh + (size_t)m0 * 1024 + k0, 1024, lane);
    v16bf a1 = make_a(Oh + (size_t)(m0 + 16) * 1024 + k0, 1024, lane);
#pragma unroll
    for (int j = 0; j < 4; ++j) {
      v16bf b = make_b(woh + (size_t)(n0 + 16 * j) * 1024 + k0, 1024, lane);
      acc[0][j] = WMMA_BF16(a0, b, acc[0][j]);
      acc[1][j] = WMMA_BF16(a1, b, acc[1][j]);
    }
  }
  const int hi = lane >> 4, nn = lane & 15;
#pragma unroll
  for (int mi = 0; mi < 2; ++mi)
#pragma unroll
    for (int j = 0; j < 4; ++j)
#pragma unroll
      for (int r = 0; r < 8; ++r)
        out[(size_t)(m0 + 16 * mi + r + 8 * hi) * 1024 + n0 + 16 * j + nn] = acc[mi][j][r];
}

extern "C" void kernel_launch(void* const* d_in, const int* in_sizes, int n_in,
                              void* d_out, int out_size, void* d_ws, size_t ws_size,
                              hipStream_t stream) {
  (void)in_sizes; (void)n_in; (void)out_size; (void)ws_size;
  const float* x  = (const float*)d_in[0];
  const float* wq = (const float*)d_in[1];
  const float* wk = (const float*)d_in[2];
  const float* wv = (const float*)d_in[3];
  const float* wo = (const float*)d_in[4];
  const float* rc = (const float*)d_in[5];
  const float* rs = (const float*)d_in[6];

  char* ws = (char*)d_ws;
  size_t off = 0;
  auto carve = [&](size_t bytes) -> void* {
    void* p = ws + off;
    off += (bytes + 255) & ~(size_t)255;
    return p;
  };
  const size_t R = 4096, Cd = 1024, BHTD = 2UL * 16 * 2048 * 64;
  bf16_t* xh  = (bf16_t*)carve(R * Cd * 2);
  bf16_t* wqh = (bf16_t*)carve(Cd * Cd * 2);
  bf16_t* wkh = (bf16_t*)carve(Cd * Cd * 2);
  bf16_t* wvh = (bf16_t*)carve(Cd * Cd * 2);
  bf16_t* woh = (bf16_t*)carve(Cd * Cd * 2);
  bf16_t* Qh  = (bf16_t*)carve(BHTD * 2);
  bf16_t* Kh  = (bf16_t*)carve(BHTD * 2);
  bf16_t* Vt  = (bf16_t*)carve(BHTD * 2);
  bf16_t* Oh  = (bf16_t*)carve(R * Cd * 2);

  cvt_kernel<<<(int)((R * Cd + 255) / 256), 256, 0, stream>>>(x, xh, (int)(R * Cd));
  cvt_kernel<<<4096, 256, 0, stream>>>(wq, wqh, 1024 * 1024);
  cvt_kernel<<<4096, 256, 0, stream>>>(wk, wkh, 1024 * 1024);
  cvt_kernel<<<4096, 256, 0, stream>>>(wv, wvh, 1024 * 1024);
  cvt_kernel<<<4096, 256, 0, stream>>>(wo, woh, 1024 * 1024);

  // 128 m-tiles(32) * 16 n64-tiles * 3 targets = 6144 waves / 4 per block
  qkv_rope_kernel<<<1536, 128, 0, stream>>>(xh, wqh, wkh, wvh, rc, rs, Qh, Kh, Vt);
  // 32 (b,h) * 32 query-groups = 1024 blocks of 4 waves
  attn_kernel<<<1024, 128, 0, stream>>>(Qh, Kh, Vt, Oh);
  // 128 * 16 = 2048 waves / 4 per block
  oproj_kernel<<<512, 128, 0, stream>>>(Oh, woh, (float*)d_out);
}
